// FullAttentionHead_2267742732849
// MI455X (gfx1250) — matrix operand order repaired and checked by reading, not compiled, not run on hardware
//
#include <hip/hip_runtime.h>
#include <hip/hip_bf16.h>

// ---------------------------------------------------------------------------
// FullAttentionHead for MI455X (gfx1250): bf16 WMMA flash attention with
// double-buffered async global->LDS staging.
// B=4, S=4096, E=1024, HEAD=64. Output fp32 [B,S,64].
// ---------------------------------------------------------------------------

typedef __attribute__((ext_vector_type(16))) __bf16 v16bf;
typedef __attribute__((ext_vector_type(8)))  __bf16 v8bf;
typedef __attribute__((ext_vector_type(8)))  float  v8f;

#define WMMA_BF16(a, b, c) \
    __builtin_amdgcn_wmma_f32_16x16x32_bf16(false, (a), false, (b), (short)0, (c), false, false)

static constexpr int Bc = 4, Sc = 4096, Ec = 1024, Hc = 64;
// softmax scale HEAD^-0.5 = 0.125 folded with log2(e) so we can use exp2f
static constexpr float QSCALE = 0.18033688011112042f;

// -------- fragment loaders ---------------------------------------------------
// B fragment (32x16, 16-bit): lane holds 16 contiguous K values.
static __device__ __forceinline__ v16bf ld_b16_contig16(const __bf16* p) {
    v8bf lo = *(const v8bf*)p;
    v8bf hi = *(const v8bf*)(p + 8);
    v16bf r;
#pragma unroll
    for (int i = 0; i < 8; ++i) { r[i] = lo[i]; r[i + 8] = hi[i]; }
    return r;
}

// A fragment (16x32, 16-bit): lane holds two 8-wide contiguous K chunks.
static __device__ __forceinline__ v16bf ld_a16_8_8(const __bf16* p0, const __bf16* p1) {
    v8bf lo = *(const v8bf*)p0;
    v8bf hi = *(const v8bf*)p1;
    v16bf r;
#pragma unroll
    for (int i = 0; i < 8; ++i) { r[i] = lo[i]; r[i + 8] = hi[i]; }
    return r;
}

// A fragment built from fp32 data (convert to bf16 on the fly).
static __device__ __forceinline__ v16bf ld_a32_8_8(const float* p0, const float* p1) {
    float4 a0 = *(const float4*)p0;
    float4 a1 = *(const float4*)(p0 + 4);
    float4 b0 = *(const float4*)p1;
    float4 b1 = *(const float4*)(p1 + 4);
    v16bf r;
    r[0]  = (__bf16)a0.x; r[1]  = (__bf16)a0.y; r[2]  = (__bf16)a0.z; r[3]  = (__bf16)a0.w;
    r[4]  = (__bf16)a1.x; r[5]  = (__bf16)a1.y; r[6]  = (__bf16)a1.z; r[7]  = (__bf16)a1.w;
    r[8]  = (__bf16)b0.x; r[9]  = (__bf16)b0.y; r[10] = (__bf16)b0.z; r[11] = (__bf16)b0.w;
    r[12] = (__bf16)b1.x; r[13] = (__bf16)b1.y; r[14] = (__bf16)b1.z; r[15] = (__bf16)b1.w;
    return r;
}

// gfx1250 async global->LDS copy, 16 bytes per lane. Tracked by ASYNCcnt.
static __device__ __forceinline__ void async_copy_b128(void* lds_dst, const void* gsrc) {
    const unsigned lds_off = (unsigned)(size_t)lds_dst;   // flat->LDS: addr[31:0]
    asm volatile("global_load_async_to_lds_b128 %0, %1, off"
                 :: "v"(lds_off), "v"(gsrc) : "memory");
}
static __device__ __forceinline__ void wait_async0() {
    asm volatile("s_wait_asynccnt 0" ::: "memory");
}

// ---------------------------------------------------------------------------
// Kernel 1: pack Wq/Wk/Wv [E,64] fp32 -> Wt [3][64][E] bf16 (transposed so B
// fragments read 16 contiguous K values per lane).
// ---------------------------------------------------------------------------
__global__ void __launch_bounds__(256) cvt_w_kernel(const float* __restrict__ Wq,
                                                    const float* __restrict__ Wk,
                                                    const float* __restrict__ Wv,
                                                    __bf16* __restrict__ Wt) {
    int idx = blockIdx.x * 256 + threadIdx.x;        // [0, 3*64*1024)
    int m   = idx >> 16;
    int rem = idx & 65535;
    int n   = rem >> 10;                             // head dim 0..63
    int e   = rem & 1023;                            // embed dim 0..1023
    const float* W = (m == 0) ? Wq : ((m == 1) ? Wk : Wv);
    Wt[idx] = (__bf16)W[e * 64 + n];
}

// ---------------------------------------------------------------------------
// Kernel 2: fused Q/K/V projection.  One wave per 16 tokens.  All 12
// B-fragments of a 32-wide K step are preloaded before the 12 WMMAs so the
// load latency overlaps with matrix issue instead of serializing.
// ---------------------------------------------------------------------------
__global__ void __launch_bounds__(256) proj_kernel(const float* __restrict__ x,
                                                   const __bf16* __restrict__ Wt,
                                                   __bf16* __restrict__ Qb,
                                                   __bf16* __restrict__ Kb,
                                                   __bf16* __restrict__ Vt) {
    const int lane = threadIdx.x & 31;
    const int wave = threadIdx.x >> 5;
    const int h = lane >> 4;        // half-wave
    const int r = lane & 15;        // row/col within fragment
    const int token_base = (blockIdx.x * 8 + wave) * 16;

    const float* xrow = x + (size_t)(token_base + r) * Ec;

    const v8f vzero = {0.f, 0.f, 0.f, 0.f, 0.f, 0.f, 0.f, 0.f};
    v8f acc[12];
#pragma unroll
    for (int mn = 0; mn < 12; ++mn) acc[mn] = vzero;

    const __bf16* wr[12];
#pragma unroll
    for (int m = 0; m < 3; ++m)
#pragma unroll
        for (int nf = 0; nf < 4; ++nf)
            wr[m * 4 + nf] = Wt + m * 65536 + (size_t)(nf * 16 + r) * Ec + 16 * h;

    for (int k = 0; k < Ec; k += 32) {
        // A fragment: X[token_base + r][k .. k+31] -> bf16
        v16bf a = ld_a32_8_8(xrow + k + 16 * h, xrow + k + 16 * h + 8);
        // preload all 12 B fragments, then issue all 12 WMMAs
        v16bf bfrag[12];
#pragma unroll
        for (int mn = 0; mn < 12; ++mn) bfrag[mn] = ld_b16_contig16(wr[mn] + k);
#pragma unroll
        for (int mn = 0; mn < 12; ++mn) acc[mn] = WMMA_BF16(a, bfrag[mn], acc[mn]);
    }

    const int bidx  = token_base >> 12;     // batch
    const int sbase = token_base & 4095;    // seq pos within batch
#pragma unroll
    for (int nf = 0; nf < 4; ++nf) {
        const int head = nf * 16 + r;       // C layout: lane column = head dim
#pragma unroll
        for (int v = 0; v < 8; ++v) {
            const int token = token_base + v + 8 * h;   // C layout row = token
            Qb[(size_t)token * 64 + head] = (__bf16)(acc[0 + nf][v] * QSCALE);
            Kb[(size_t)token * 64 + head] = (__bf16)(acc[4 + nf][v]);
        }
        // V transposed: lane column (head) becomes a row -> contiguous store
        v8bf pv;
#pragma unroll
        for (int v = 0; v < 8; ++v) pv[v] = (__bf16)acc[8 + nf][v];
        *(v8bf*)(Vt + ((size_t)(bidx * 64 + head)) * Sc + sbase + 8 * h) = pv;
    }
}

// ---------------------------------------------------------------------------
// Kernel 3: flash attention.  8 waves/WG, 16 queries per wave (128/WG).
// Double-buffered async global->LDS staging of 128-key K / V^T tiles
// (2 x 32 KB LDS): prefetch tile kt+1 with global_load_async_to_lds_b128
// while computing tile kt, then s_wait_asynccnt 0 + barrier.
// Per 32-key step per wave: 4 WMMA (S^T = K*Q^T) + 4 WMMA (O^T += V^T*P^T).
// ---------------------------------------------------------------------------
__global__ void __launch_bounds__(256) attn_kernel(const __bf16* __restrict__ Qb,
                                                   const __bf16* __restrict__ Kb,
                                                   const __bf16* __restrict__ Vt,
                                                   float* __restrict__ out) {
    // double buffer: [buf][ K tile 16KB | V^T tile 16KB ]
    __shared__ __align__(16) char smem[2 * 32768];

    const int t    = threadIdx.x;
    const int lane = t & 31;
    const int wave = t >> 5;
    const int h = lane >> 4;
    const int r = lane & 15;

    const int bidx   = blockIdx.x >> 5;                  // batch
    const int q_base = (blockIdx.x & 31) * 128 + wave * 16;

    const __bf16* kbase = Kb + (size_t)bidx * Sc * 64;
    const __bf16* vbase = Vt + (size_t)bidx * 64 * Sc;

    // Q^T B fragments (K-dim = head dims), loaded once. Q already scaled.
    const __bf16* qrow = Qb + (size_t)(bidx * Sc + q_base + r) * 64;
    const v16bf bq0 = ld_a16_8_8(qrow + 8 * h, qrow + 16 + 8 * h);        // heads 0..31
    const v16bf bq1 = ld_a16_8_8(qrow + 32 + 8 * h, qrow + 48 + 8 * h);   // heads 32..63

    const v8f vzero = {0.f, 0.f, 0.f, 0.f, 0.f, 0.f, 0.f, 0.f};
    v8f o[4];
#pragma unroll
    for (int f = 0; f < 4; ++f) o[f] = vzero;
    float mrun = -__builtin_inff();
    float lrun = 0.f;

    // issue async loads for key-tile kt into buffer buf (8 x b128 per thread)
    auto issue_tile = [&](int kt, int buf) {
        char* dstK = smem + buf * 32768;
        char* dstV = smem + buf * 32768 + 16384;
        const char* srcK = (const char*)kbase + (size_t)kt * 16384;   // 128*64*2
#pragma unroll
        for (int i = 0; i < 4; ++i) {
            const int c = t + 256 * i;                                // 16B chunk id
            async_copy_b128(dstK + c * 16, srcK + c * 16);
        }
#pragma unroll
        for (int i = 0; i < 4; ++i) {
            const int c = t + 256 * i;
            const int head = c >> 4, off = c & 15;
            const char* g = (const char*)vbase + (size_t)head * (Sc * 2)
                          + (size_t)kt * 256 + off * 16;
            async_copy_b128(dstV + c * 16, g);
        }
    };

    constexpr int NT = Sc / 128;    // 32 key tiles
    issue_tile(0, 0);
    wait_async0();
    __syncthreads();

    for (int kt = 0; kt < NT; ++kt) {
        const int cur = kt & 1;
        if (kt + 1 < NT) issue_tile(kt + 1, cur ^ 1);   // prefetch next tile

        const __bf16* sK = (const __bf16*)(smem + cur * 32768);           // [key][head]
        const __bf16* sV = (const __bf16*)(smem + cur * 32768 + 16384);   // [head][key]

#pragma unroll
        for (int st = 0; st < 4; ++st) {
            const int key0 = st * 32;

            // ---- S^T = K * Q^T for two 16-key subtiles -------------------
            const __bf16* krow0 = sK + (size_t)(key0 + r) * 64;
            const __bf16* krow1 = sK + (size_t)(key0 + 16 + r) * 64;
            v8f s0 = vzero, s1 = vzero;
            s0 = WMMA_BF16(ld_a16_8_8(krow0 + 8 * h, krow0 + 16 + 8 * h), bq0, s0);
            s0 = WMMA_BF16(ld_a16_8_8(krow0 + 32 + 8 * h, krow0 + 48 + 8 * h), bq1, s0);
            s1 = WMMA_BF16(ld_a16_8_8(krow1 + 8 * h, krow1 + 16 + 8 * h), bq0, s1);
            s1 = WMMA_BF16(ld_a16_8_8(krow1 + 32 + 8 * h, krow1 + 48 + 8 * h), bq1, s1);

            // ---- online softmax (values already in log2 domain) ----------
            float tmax = -__builtin_inff();
#pragma unroll
            for (int v = 0; v < 8; ++v) tmax = fmaxf(tmax, fmaxf(s0[v], s1[v]));
            tmax = fmaxf(tmax, __shfl_xor(tmax, 16, 32));
            const float mnew  = fmaxf(mrun, tmax);
            const float alpha = exp2f(mrun - mnew);

            float p0[8], p1[8];
            float psum = 0.f;
#pragma unroll
            for (int v = 0; v < 8; ++v) {
                p0[v] = exp2f(s0[v] - mnew);
                p1[v] = exp2f(s1[v] - mnew);
                psum += p0[v] + p1[v];
            }
            psum += __shfl_xor(psum, 16, 32);
            lrun = lrun * alpha + psum;
            mrun = mnew;
#pragma unroll
            for (int f = 0; f < 4; ++f) o[f] *= alpha;

            // ---- build P^T B fragment (half-wave exchange) ---------------
            v16bf bp;
#pragma unroll
            for (int v = 0; v < 8; ++v) {
                const float p0p = __shfl_xor(p0[v], 16, 32);
                const float p1p = __shfl_xor(p1[v], 16, 32);
                bp[v]     = (__bf16)(h ? p1p   : p0[v]);
                bp[8 + v] = (__bf16)(h ? p1[v] : p0p);
            }

            // ---- O^T += V^T * P^T ----------------------------------------
#pragma unroll
            for (int f = 0; f < 4; ++f) {
                const __bf16* vrow = sV + (size_t)(f * 16 + r) * 128 + key0;
                v16bf av = ld_a16_8_8(vrow + 16 * h, vrow + 16 * h + 8);
                o[f] = WMMA_BF16(av, bp, o[f]);
            }
        }

        wait_async0();      // prefetch for kt+1 has landed in LDS
        __syncthreads();    // all waves done reading buf cur / writing cur^1
    }

    // ---- epilogue: normalize and store fp32 [B,S,64] -----------------------
    const float rl = 1.0f / lrun;
    float* op = out + (size_t)(bidx * Sc + q_base + r) * 64;
#pragma unroll
    for (int f = 0; f < 4; ++f) {
        float4 lo4 = make_float4(o[f][0] * rl, o[f][1] * rl, o[f][2] * rl, o[f][3] * rl);
        float4 hi4 = make_float4(o[f][4] * rl, o[f][5] * rl, o[f][6] * rl, o[f][7] * rl);
        *(float4*)(op + f * 16 + 8 * h)     = lo4;
        *(float4*)(op + f * 16 + 8 * h + 4) = hi4;
    }
}

// ---------------------------------------------------------------------------
extern "C" void kernel_launch(void* const* d_in, const int* in_sizes, int n_in,
                              void* d_out, int out_size, void* d_ws, size_t ws_size,
                              hipStream_t stream) {
    (void)in_sizes; (void)n_in; (void)out_size; (void)ws_size;

    const float* x  = (const float*)d_in[0];
    const float* Wq = (const float*)d_in[1];
    const float* Wk = (const float*)d_in[2];
    const float* Wv = (const float*)d_in[3];
    float* out = (float*)d_out;

    // workspace layout (bytes)
    char* ws = (char*)d_ws;
    __bf16* Wt = (__bf16*)(ws);                                  // 3*64*1024*2 = 393216
    __bf16* Qb = (__bf16*)(ws + 393216);                         // 2 MB
    __bf16* Kb = (__bf16*)(ws + 393216 + 2097152);               // 2 MB
    __bf16* Vt = (__bf16*)(ws + 393216 + 2 * 2097152);           // 2 MB

    cvt_w_kernel<<<768, 256, 0, stream>>>(Wq, Wk, Wv, Wt);
    proj_kernel<<<(Bc * Sc) / (8 * 16), 256, 0, stream>>>(x, Wt, Qb, Kb, Vt);
    attn_kernel<<<Bc * (Sc / 128), 256, 0, stream>>>(Qb, Kb, Vt, out);
}
